// PredRNN_27350351740967
// MI455X (gfx1250) — compile-verified, matrix operand order checked by reading, not run on hardware
//
#include <hip/hip_runtime.h>
#include <hip/hip_bf16.h>

// ---------------------------------------------------------------------------
// PredRNN (ST-LSTM) for MI455X / gfx1250.
// Conv layers as implicit-GEMM via v_wmma_f32_16x16x32_bf16 (wave32 WMMA).
// Register-blocked 64x64 (MTxNT) output tiles per wave: 1.0 vmem per wmma.
// Activations: NHWC bf16 (channel-contiguous, padded to multiple of 32).
// Weights: repacked to [Cout][kpos][Cin] bf16 once per launch.
// Gates: f32 elementwise kernels.
// ---------------------------------------------------------------------------

typedef __bf16 bf16_t;
typedef __attribute__((ext_vector_type(16))) __bf16 v16bf;
typedef __attribute__((ext_vector_type(8)))  __bf16 v8bf;
typedef __attribute__((ext_vector_type(8)))  float  v8f;

#define BB    16
#define TT    20
#define CCH   16
#define HH    32
#define WWD   32
#define LLAY  4
#define NHID  64
#define KSZ   5
#define INLEN 10
#define NPIX  (BB * HH * WWD)      // 16384 = B*H*W "pixel" GEMM-N dimension

// ---------------------------------------------------------------------------
// Implicit-GEMM conv: out[Cout][NPIX] (f32) = Wpack[Cout][KPOS][Cpad] x
// im2col(in NHWC-bf16). One wave per block computing a (16*MT) x (16*NT)
// output tile: A fragments reused NT times, B fragments reused MT times.
// ---------------------------------------------------------------------------
template <int MT, int NT>
__global__ __launch_bounds__(32)
void conv_wmma_kernel(const bf16_t* __restrict__ in,   // NHWC, stride Cpad
                      const bf16_t* __restrict__ wp,   // [Cout][KPOS][Cpad]
                      float* __restrict__ out,         // [Cout][NPIX]
                      int Cpad, int KS, int pad) {
    const int lane = threadIdx.x;
    const int g    = lane >> 4;      // K-half selector (lanes 0-15 vs 16-31)
    const int lm   = lane & 15;      // row (A) / column (B) within tile
    const int n0   = blockIdx.x * (16 * NT);
    const int cout0 = blockIdx.y * (16 * MT);
    const int KPOS  = KS * KS;

    int pb[NT], py[NT], px[NT];
#pragma unroll
    for (int nt = 0; nt < NT; ++nt) {
        int n = n0 + nt * 16 + lm;
        pb[nt] = n >> 10;
        int rem = n & 1023;
        py[nt] = rem >> 5;
        px[nt] = rem & 31;
    }

    v8f acc[MT][NT];
#pragma unroll
    for (int mt = 0; mt < MT; ++mt)
#pragma unroll
        for (int nt = 0; nt < NT; ++nt)
#pragma unroll
            for (int r = 0; r < 8; ++r) acc[mt][nt][r] = 0.0f;

    for (int kh = 0; kh < KS; ++kh) {
        for (int kw = 0; kw < KS; ++kw) {
            const int kpos = kh * KS + kw;
            bool valid[NT];
            const bf16_t* bbase[NT];
#pragma unroll
            for (int nt = 0; nt < NT; ++nt) {
                int yy = py[nt] + kh - pad;
                int xx = px[nt] + kw - pad;
                valid[nt] = ((unsigned)yy < (unsigned)HH) &&
                            ((unsigned)xx < (unsigned)WWD);
                bbase[nt] = in + ((size_t)((pb[nt] * HH + yy) * WWD + xx)) * Cpad;
            }
            for (int cb = 0; cb < Cpad; cb += 32) {
                // A fragments: 16(M) x 32(K), row = contiguous packed K
                v16bf afrag[MT];
#pragma unroll
                for (int mt = 0; mt < MT; ++mt) {
                    const bf16_t* abase =
                        wp + ((size_t)((cout0 + mt * 16 + lm) * KPOS + kpos)) * Cpad + cb;
                    v8bf alo = *(const v8bf*)(abase + g * 8);
                    v8bf ahi = *(const v8bf*)(abase + 16 + g * 8);
                    afrag[mt] = __builtin_shufflevector(alo, ahi,
                        0,1,2,3,4,5,6,7,8,9,10,11,12,13,14,15);
                }
#pragma unroll
                for (int nt = 0; nt < NT; ++nt) {
                    // B fragment: 32(K) x 16(N), column = contiguous channels
                    v16bf bfrag;
                    if (valid[nt]) {
                        v8bf blo = *(const v8bf*)(bbase[nt] + cb + g * 8);
                        v8bf bhi = *(const v8bf*)(bbase[nt] + cb + 16 + g * 8);
                        bfrag = __builtin_shufflevector(blo, bhi,
                            0,1,2,3,4,5,6,7,8,9,10,11,12,13,14,15);
                    } else {
#pragma unroll
                        for (int j = 0; j < 16; ++j) bfrag[j] = (bf16_t)0.0f;
                    }
#pragma unroll
                    for (int mt = 0; mt < MT; ++mt) {
                        acc[mt][nt] = __builtin_amdgcn_wmma_f32_16x16x32_bf16(
                            false, afrag[mt], false, bfrag,
                            (short)0, acc[mt][nt], false, false);
                    }
                }
            }
        }
    }
    // C/D layout: VGPR r, lanes 0-15 -> M=r, lanes 16-31 -> M=r+8; N=lane&15
#pragma unroll
    for (int mt = 0; mt < MT; ++mt)
#pragma unroll
        for (int nt = 0; nt < NT; ++nt)
#pragma unroll
            for (int r = 0; r < 8; ++r) {
                int row = cout0 + mt * 16 + g * 8 + r;
                out[(size_t)row * NPIX + n0 + nt * 16 + lm] = acc[mt][nt][r];
            }
}

// ---------------------------------------------------------------------------
// Weight repack: [Cout][Cin][kh][kw] f32 -> [Cout][KPOS][Cpad] bf16 (0-padded)
// ---------------------------------------------------------------------------
__global__ void pack_w_kernel(const float* __restrict__ w, bf16_t* __restrict__ wp,
                              int Cout, int Cin, int KPOS, int Cpad) {
    int idx = blockIdx.x * blockDim.x + threadIdx.x;
    int total = Cout * KPOS * Cpad;
    if (idx >= total) return;
    int cp   = idx % Cpad;
    int tmp  = idx / Cpad;
    int kpos = tmp % KPOS;
    int co   = tmp / KPOS;
    float v = (cp < Cin) ? w[((size_t)co * Cin + cp) * KPOS + kpos] : 0.0f;
    wp[idx] = (bf16_t)v;
}

__device__ __forceinline__ float sigmoidf_(float v) {
    return 1.0f / (1.0f + __expf(-v));
}

// net = mask*frame + (1-mask)*x_gen, stored NHWC bf16 padded to 32 channels
__global__ void net_kernel(const float* __restrict__ frames,
                           const float* __restrict__ mask_true,
                           const float* __restrict__ xgen,  // [C][NPIX] f32
                           bf16_t* __restrict__ net, int t) {
    int idx = blockIdx.x * blockDim.x + threadIdx.x;
    if (idx >= NPIX * 32) return;
    int c = idx & 31;
    int n = idx >> 5;
    float v = 0.0f;
    if (c < CCH) {
        int b = n >> 10, rem = n & 1023, y = rem >> 5, x = rem & 31;
        float f = frames[((((size_t)b * TT + t) * CCH + c) * HH + y) * WWD + x];
        float m = 1.0f;
        if (t >= INLEN)
            m = mask_true[((((size_t)b * (TT - INLEN - 1) + (t - INLEN)) * CCH + c) * HH + y) * WWD + x];
        float xg = xgen[(size_t)c * NPIX + n];
        v = m * f + (1.0f - m) * xg;
    }
    net[(size_t)n * 32 + c] = (bf16_t)v;
}

// Gate stage 1: i/f/g (temporal) + i'/f'/g' (spatiotemporal) -> c_new, m_new,
// plus bf16 NHWC packs of mem=[c_new;m_new] (128ch) and m_new (64ch).
__global__ void gate1_kernel(const float* __restrict__ xc,  // [7*64][NPIX]
                             const float* __restrict__ hc,  // [4*64][NPIX]
                             const float* __restrict__ mc,  // [3*64][NPIX]
                             float* __restrict__ c_st,      // [64][NPIX] in/out
                             float* __restrict__ m_st,      // [64][NPIX] in/out
                             bf16_t* __restrict__ mem_bf,   // NHWC 128
                             bf16_t* __restrict__ m_bf) {   // NHWC 64
    int idx = blockIdx.x * blockDim.x + threadIdx.x;
    if (idx >= 64 * NPIX) return;
    int n  = idx & (NPIX - 1);
    int ch = idx >> 14;
    size_t o = (size_t)ch * NPIX + n;
    float i_t = sigmoidf_(xc[o] + hc[o]);
    float f_t = sigmoidf_(xc[(size_t)(64 + ch) * NPIX + n] +
                          hc[(size_t)(64 + ch) * NPIX + n] + 1.0f);
    float g_t = tanhf(xc[(size_t)(128 + ch) * NPIX + n] +
                      hc[(size_t)(128 + ch) * NPIX + n]);
    float cn = f_t * c_st[o] + i_t * g_t;
    float i_p = sigmoidf_(xc[(size_t)(192 + ch) * NPIX + n] + mc[o]);
    float f_p = sigmoidf_(xc[(size_t)(256 + ch) * NPIX + n] +
                          mc[(size_t)(64 + ch) * NPIX + n] + 1.0f);
    float g_p = tanhf(xc[(size_t)(320 + ch) * NPIX + n] +
                      mc[(size_t)(128 + ch) * NPIX + n]);
    float mn = f_p * m_st[o] + i_p * g_p;
    c_st[o] = cn;
    m_st[o] = mn;
    mem_bf[(size_t)n * 128 + ch]      = (bf16_t)cn;
    mem_bf[(size_t)n * 128 + 64 + ch] = (bf16_t)mn;
    m_bf[(size_t)n * 64 + ch]         = (bf16_t)mn;
}

// Gate stage 2: o_t = sigmoid(o_x + o_h + oc); h = o_t * tanh(lc) -> bf16 NHWC
__global__ void gate2_kernel(const float* __restrict__ xc,
                             const float* __restrict__ hc,
                             const float* __restrict__ oc,  // [64][NPIX]
                             const float* __restrict__ lc,  // [64][NPIX]
                             bf16_t* __restrict__ h_bf) {   // NHWC 64
    int idx = blockIdx.x * blockDim.x + threadIdx.x;
    if (idx >= 64 * NPIX) return;
    int n  = idx & (NPIX - 1);
    int ch = idx >> 14;
    size_t o = (size_t)ch * NPIX + n;
    float o_t = sigmoidf_(xc[(size_t)(384 + ch) * NPIX + n] +
                          hc[(size_t)(192 + ch) * NPIX + n] + oc[o]);
    float hn = o_t * tanhf(lc[o]);
    h_bf[(size_t)n * 64 + ch] = (bf16_t)hn;
}

// Scatter x_gen [C][NPIX] -> d_out [B][T-1][C][H][W]
__global__ void emit_kernel(const float* __restrict__ xgen,
                            float* __restrict__ out, int t) {
    int idx = blockIdx.x * blockDim.x + threadIdx.x;
    if (idx >= CCH * NPIX) return;
    int n  = idx & (NPIX - 1);
    int ch = idx >> 14;
    int b  = n >> 10;
    int p  = n & 1023;
    out[((((size_t)b * (TT - 1) + t) * CCH + ch) << 10) + p] = xgen[idx];
}

// ---------------------------------------------------------------------------
extern "C" void kernel_launch(void* const* d_in, const int* in_sizes, int n_in,
                              void* d_out, int out_size, void* d_ws, size_t ws_size,
                              hipStream_t stream) {
    const float* frames    = (const float*)d_in[0];
    const float* mask_true = (const float*)d_in[1];
    const float* Wx0   = (const float*)d_in[2];
    const float* Wxr   = (const float*)d_in[3];
    const float* Wh    = (const float*)d_in[4];
    const float* Wm    = (const float*)d_in[5];
    const float* Wo    = (const float*)d_in[6];
    const float* Wl    = (const float*)d_in[7];
    const float* Wlast = (const float*)d_in[8];
    float* out = (float*)d_out;

    // ---- carve workspace (256B aligned slices) ----
    char* base = (char*)d_ws;
    size_t off = 0;
    auto carve = [&](size_t bytes) -> void* {
        void* p = base + off;
        off = (off + bytes + 255) & ~(size_t)255;
        return p;
    };
    float*  xc    = (float*)carve((size_t)448 * NPIX * 4);
    float*  hc    = (float*)carve((size_t)256 * NPIX * 4);
    float*  mc    = (float*)carve((size_t)192 * NPIX * 4);
    float*  oc    = (float*)carve((size_t)64  * NPIX * 4);
    float*  lc    = (float*)carve((size_t)64  * NPIX * 4);
    float*  xgen  = (float*)carve((size_t)CCH * NPIX * 4);
    bf16_t* netb  = (bf16_t*)carve((size_t)NPIX * 32 * 2);
    bf16_t* hbf   = (bf16_t*)carve((size_t)LLAY * NPIX * 64 * 2);
    float*  cst   = (float*)carve((size_t)LLAY * 64 * NPIX * 4);
    float*  mst   = (float*)carve((size_t)64 * NPIX * 4);
    bf16_t* mbf   = (bf16_t*)carve((size_t)NPIX * 64 * 2);
    bf16_t* membf = (bf16_t*)carve((size_t)NPIX * 128 * 2);
    bf16_t* wx0p  = (bf16_t*)carve((size_t)448 * 25 * 32 * 2);
    bf16_t* wxrp  = (bf16_t*)carve((size_t)3 * 448 * 25 * 64 * 2);
    bf16_t* whp   = (bf16_t*)carve((size_t)4 * 256 * 25 * 64 * 2);
    bf16_t* wmp   = (bf16_t*)carve((size_t)4 * 192 * 25 * 64 * 2);
    bf16_t* wop   = (bf16_t*)carve((size_t)4 * 64 * 25 * 128 * 2);
    bf16_t* wlp   = (bf16_t*)carve((size_t)4 * 64 * 128 * 2);
    bf16_t* wlastp= (bf16_t*)carve((size_t)16 * 64 * 2);

    // ---- zero-init recurrent state + xgen (graph-capture-safe memsets) ----
    hipMemsetAsync(xgen, 0, (size_t)CCH * NPIX * 4, stream);
    hipMemsetAsync(hbf,  0, (size_t)LLAY * NPIX * 64 * 2, stream);
    hipMemsetAsync(cst,  0, (size_t)LLAY * 64 * NPIX * 4, stream);
    hipMemsetAsync(mst,  0, (size_t)64 * NPIX * 4, stream);
    hipMemsetAsync(mbf,  0, (size_t)NPIX * 64 * 2, stream);

    // ---- repack weights to bf16 [Cout][kpos][Cin] ----
    auto pack = [&](const float* w, bf16_t* wp, int Cout, int Cin, int KPOS, int Cpad) {
        int total = Cout * KPOS * Cpad;
        pack_w_kernel<<<(total + 255) / 256, 256, 0, stream>>>(w, wp, Cout, Cin, KPOS, Cpad);
    };
    pack(Wx0, wx0p, 448, 16, 25, 32);
    for (int l = 0; l < 3; ++l)
        pack(Wxr + (size_t)l * 448 * 64 * 25, wxrp + (size_t)l * 448 * 25 * 64, 448, 64, 25, 64);
    for (int l = 0; l < 4; ++l) {
        pack(Wh + (size_t)l * 256 * 64 * 25, whp + (size_t)l * 256 * 25 * 64, 256, 64, 25, 64);
        pack(Wm + (size_t)l * 192 * 64 * 25, wmp + (size_t)l * 192 * 25 * 64, 192, 64, 25, 64);
        pack(Wo + (size_t)l * 64 * 128 * 25, wop + (size_t)l * 64 * 25 * 128, 64, 128, 25, 128);
        pack(Wl + (size_t)l * 64 * 128,      wlp + (size_t)l * 64 * 128,      64, 128, 1, 128);
    }
    pack(Wlast, wlastp, 16, 64, 1, 64);

    const dim3 wv(32, 1, 1);
    // ---- time loop ----
    for (int t = 0; t < TT - 1; ++t) {
        net_kernel<<<(NPIX * 32 + 255) / 256, 256, 0, stream>>>(
            frames, mask_true, xgen, netb, t);

        for (int l = 0; l < LLAY; ++l) {
            const bf16_t* xin;
            const bf16_t* wxp;
            int cpad_x;
            if (l == 0) { xin = netb; wxp = wx0p; cpad_x = 32; }
            else {
                xin = hbf + (size_t)(l - 1) * NPIX * 64;
                wxp = wxrp + (size_t)(l - 1) * 448 * 25 * 64;
                cpad_x = 64;
            }
            // xc = conv5x5(x, Wx): 448 out, 64x64 tiles
            conv_wmma_kernel<4, 4><<<dim3(NPIX / 64, 448 / 64), wv, 0, stream>>>(
                xin, wxp, xc, cpad_x, KSZ, KSZ / 2);
            // hc = conv5x5(h, Wh): 256 out
            conv_wmma_kernel<4, 4><<<dim3(NPIX / 64, 256 / 64), wv, 0, stream>>>(
                hbf + (size_t)l * NPIX * 64, whp + (size_t)l * 256 * 25 * 64,
                hc, 64, KSZ, KSZ / 2);
            // mc = conv5x5(m, Wm): 192 out
            conv_wmma_kernel<4, 4><<<dim3(NPIX / 64, 192 / 64), wv, 0, stream>>>(
                mbf, wmp + (size_t)l * 192 * 25 * 64, mc, 64, KSZ, KSZ / 2);

            gate1_kernel<<<(64 * NPIX + 255) / 256, 256, 0, stream>>>(
                xc, hc, mc, cst + (size_t)l * 64 * NPIX, mst, membf, mbf);

            // oc = conv5x5(mem, Wo): 64 out over 128 in; 64x32 tiles (parallelism)
            conv_wmma_kernel<4, 2><<<dim3(NPIX / 32, 1), wv, 0, stream>>>(
                membf, wop + (size_t)l * 64 * 25 * 128, oc, 128, KSZ, KSZ / 2);
            // lc = conv1x1(mem, Wl): 64 out over 128 in
            conv_wmma_kernel<4, 2><<<dim3(NPIX / 32, 1), wv, 0, stream>>>(
                membf, wlp + (size_t)l * 64 * 128, lc, 128, 1, 0);

            gate2_kernel<<<(64 * NPIX + 255) / 256, 256, 0, stream>>>(
                xc, hc, oc, lc, hbf + (size_t)l * NPIX * 64);
        }
        // x_gen = conv1x1(h[L-1], Wlast): 16 out over 64 in
        conv_wmma_kernel<1, 4><<<dim3(NPIX / 64, 1), wv, 0, stream>>>(
            hbf + (size_t)(LLAY - 1) * NPIX * 64, wlastp, xgen, 64, 1, 0);
        emit_kernel<<<(CCH * NPIX + 255) / 256, 256, 0, stream>>>(xgen, out, t);
    }
}
